// HBS_73882027425848
// MI455X (gfx1250) — matrix operand order repaired and checked by reading, not compiled, not run on hardware
//
#include <hip/hip_runtime.h>
#include <hip/hip_bf16.h>

#define NODES 100000
#define NEDGE 1000000
#define CF 64
#define EPSV 1e-5f
#define NTILES (NODES / 16)   // 6250 exact
#define EPT 16                // edges per thread (run-length compression window)

typedef __attribute__((ext_vector_type(16))) _Float16 v16h;
typedef __attribute__((ext_vector_type(8)))  float    v8f;

__device__ __forceinline__ v8f wmma16(v16h a, v16h b, v8f c) {
  // D = A(16x32 f16) * B(32x16 f16) + C(16x16 f32)
  return __builtin_amdgcn_wmma_f32_16x16x32_f16(
      /*neg_a=*/false, a, /*neg_b=*/false, b,
      /*c_mod=*/(short)0, c, /*reuse_a=*/false, /*reuse_b=*/false);
}

// ---------------------------------------------------------------------------
// Pre-pack a KxN(=64) f32 weight matrix into per-lane WMMA B-fragment order:
// fragment (kt, nt), lane L holds 16 f16 = W[kt*32+(L>>4)*8 + (j<8 ? j : j+8)]
// [nt*16 + (L&15)], j=0..15.  One 32B load per fragment afterwards, no cvt.
// ---------------------------------------------------------------------------
__global__ void __launch_bounds__(128) pna_pack_b(
    const float* __restrict__ W, _Float16* __restrict__ out, int K) {
  const int t = blockIdx.x * blockDim.x + threadIdx.x;
  const int ntiles = (K >> 5) * 4;           // (K/32) k-tiles x 4 n-tiles
  if (t >= ntiles * 32) return;
  const int lane = t & 31;
  const int tile = t >> 5;                   // tile = kt*4 + nt
  const int nt = tile & 3;
  const int kt = tile >> 2;
  const int n  = nt * 16 + (lane & 15);
  const int kb = kt * 32 + (lane >> 4) * 8;
  _Float16* dst = out + (size_t)t * 16;
#pragma unroll
  for (int j = 0; j < 16; ++j)
    dst[j] = (_Float16)W[(size_t)(kb + (j < 8 ? j : j + 8)) * CF + n];
}

// ---------------------------------------------------------------------------
// Zero / init the per-node stat accumulators (sum, sumsq, max, count).
// ---------------------------------------------------------------------------
__global__ void __launch_bounds__(256) pna_init_stats(
    float* __restrict__ s, float* __restrict__ q,
    float* __restrict__ m, float* __restrict__ c) {
  const int i = blockIdx.x * blockDim.x + threadIdx.x;
  if (i < NODES * CF) { s[i] = 0.f; q[i] = 0.f; m[i] = -3.402823466e38f; }
  if (i < NODES) c[i] = 0.f;
}

// ---------------------------------------------------------------------------
// message  = x @ w1 ; message2 = x @ w2   (WMMA, one wave / 16 rows,
// pre-packed f16 B fragments: pure load -> wmma inner loop)
// ---------------------------------------------------------------------------
__global__ void __launch_bounds__(256) pna_msg_gemm(
    const float* __restrict__ x, const _Float16* __restrict__ wh1,
    const _Float16* __restrict__ wh2, float* __restrict__ m1,
    float* __restrict__ m2) {
  const int wid = blockIdx.x * (blockDim.x >> 5) + (threadIdx.x >> 5);
  if (wid >= NTILES) return;
  const int lane = threadIdx.x & 31;
  const int row0 = wid * 16;

  // A fragment layout (16-bit A 16x32): lane<16 -> M=lane, K groups {0..7,16..23};
  // lane>=16 -> M=lane-16, K groups {8..15,24..31}.
  const int am  = lane & 15;
  const int akb = (lane >> 4) * 8;
  const float* xr = x + (size_t)(row0 + am) * CF;
  v16h a0, a1;
#pragma unroll
  for (int v = 0; v < 8; ++v) {
    const int k = (v < 4 ? 2 * v : 16 + 2 * (v - 4)) + akb;
    a0[2 * v]     = (_Float16)xr[k];
    a0[2 * v + 1] = (_Float16)xr[k + 1];
    a1[2 * v]     = (_Float16)xr[32 + k];
    a1[2 * v + 1] = (_Float16)xr[32 + k + 1];
  }

#pragma unroll
  for (int nt = 0; nt < 4; ++nt) {
    const v16h p0 = *(const v16h*)(wh1 + ((size_t)(0 * 4 + nt) * 32 + lane) * 16);
    const v16h p1 = *(const v16h*)(wh1 + ((size_t)(1 * 4 + nt) * 32 + lane) * 16);
    const v16h q0 = *(const v16h*)(wh2 + ((size_t)(0 * 4 + nt) * 32 + lane) * 16);
    const v16h q1 = *(const v16h*)(wh2 + ((size_t)(1 * 4 + nt) * 32 + lane) * 16);
    v8f c1 = {}; c1 = wmma16(a0, p0, c1); c1 = wmma16(a1, p1, c1);
    v8f c2 = {}; c2 = wmma16(a0, q0, c2); c2 = wmma16(a1, q1, c2);
    const int ncol = nt * 16 + (lane & 15);
#pragma unroll
    for (int v = 0; v < 8; ++v) {
      const int mm = v + (lane >> 4) * 8;   // C/D layout: M = v (+8 for hi half)
      m1[(size_t)(row0 + mm) * CF + ncol] = c1[v];
      m2[(size_t)(row0 + mm) * CF + ncol] = c2[v];
    }
  }
}

// ---------------------------------------------------------------------------
// Edge aggregation: run-length compressed atomics (rows are sorted).
// Thread = (channel-group of 4 floats, chunk of EPT consecutive edges).
// ---------------------------------------------------------------------------
__device__ __forceinline__ void pna_flush(
    float* __restrict__ s, float* __restrict__ q, float* __restrict__ m,
    float* __restrict__ cnt, int node, int cg,
    const float* as, const float* aq, const float* am, float count) {
  const size_t base = (size_t)node * CF + cg * 4;
#pragma unroll
  for (int i = 0; i < 4; ++i) {
    atomicAdd(&s[base + i], as[i]);
    atomicAdd(&q[base + i], aq[i]);
    __hip_atomic_fetch_max(&m[base + i], am[i], __ATOMIC_RELAXED,
                           __HIP_MEMORY_SCOPE_AGENT);
  }
  if (cg == 0) atomicAdd(&cnt[node], count);
}

__global__ void __launch_bounds__(256) pna_edge_stats(
    const int* __restrict__ erow, const int* __restrict__ ecol,
    const float* __restrict__ msg, float* __restrict__ s,
    float* __restrict__ q, float* __restrict__ m, float* __restrict__ cnt) {
  const int t = blockIdx.x * blockDim.x + threadIdx.x;
  const int cg = t & 15;        // 4 channels each; 16 lanes cover a full row
  const int chunk = t >> 4;
  const int e0 = chunk * EPT;
  if (e0 >= NEDGE) return;
  const int e1 = min(e0 + EPT, NEDGE);

  int cur = erow[e0];
  float as[4] = {0.f, 0.f, 0.f, 0.f};
  float aq[4] = {0.f, 0.f, 0.f, 0.f};
  float am[4] = {-3.402823466e38f, -3.402823466e38f,
                 -3.402823466e38f, -3.402823466e38f};
  float count = 0.f;

  for (int e = e0; e < e1; ++e) {
    const int r = erow[e];
    if (r != cur) {
      pna_flush(s, q, m, cnt, cur, cg, as, aq, am, count);
#pragma unroll
      for (int i = 0; i < 4; ++i) { as[i] = 0.f; aq[i] = 0.f; am[i] = -3.402823466e38f; }
      count = 0.f;
      cur = r;
    }
    const float4 g =
        *reinterpret_cast<const float4*>(msg + (size_t)ecol[e] * CF + cg * 4);
    as[0] += g.x; as[1] += g.y; as[2] += g.z; as[3] += g.w;
    aq[0] += g.x * g.x; aq[1] += g.y * g.y; aq[2] += g.z * g.z; aq[3] += g.w * g.w;
    am[0] = fmaxf(am[0], g.x); am[1] = fmaxf(am[1], g.y);
    am[2] = fmaxf(am[2], g.z); am[3] = fmaxf(am[3], g.w);
    count += 1.f;
  }
  pna_flush(s, q, m, cnt, cur, cg, as, aq, am, count);
}

// ---------------------------------------------------------------------------
// Fused finalize: stats -> feats(fp16, LDS) -> WMMA GEMM (both aggregations
// accumulated, pre-packed B) -> +bias -> relu -> LayerNorm -> out.
// One wave / 16 nodes.
// ---------------------------------------------------------------------------
__global__ void __launch_bounds__(128) pna_finalize(
    const float* __restrict__ s1, const float* __restrict__ q1,
    const float* __restrict__ m1, const float* __restrict__ c1,
    const float* __restrict__ s2, const float* __restrict__ q2,
    const float* __restrict__ m2, const float* __restrict__ c2,
    const _Float16* __restrict__ wh1, const float* __restrict__ b1,
    const _Float16* __restrict__ wh2, const float* __restrict__ b2,
    const float* __restrict__ gamma, const float* __restrict__ beta,
    float* __restrict__ out) {
  __shared__ _Float16 sfeat[4][16 * 256];  // per-wave feats tile (8 KB each)
  __shared__ float    sres[4][16 * 64];    // per-wave pre-LN result (4 KB each)

  const int wid  = threadIdx.x >> 5;
  const int lane = threadIdx.x & 31;
  int tile = blockIdx.x * 4 + wid;
  if (tile >= NTILES) tile = NTILES - 1;   // duplicate work, identical writes
  const int row0 = tile * 16;

  const int am  = lane & 15;
  const int akb = (lane >> 4) * 8;
  const int bn  = lane & 15;

  v8f acc[4];
#pragma unroll
  for (int nt = 0; nt < 4; ++nt) acc[nt] = (v8f){};

  for (int agg = 0; agg < 2; ++agg) {
    const float* S = agg ? s2 : s1;
    const float* Q = agg ? q2 : q1;
    const float* M = agg ? m2 : m1;
    const float* C = agg ? c2 : c1;
    const _Float16* Wh = agg ? wh2 : wh1;

    // Build feats = [sum | mean | max | std] for 16 nodes into LDS (fp16).
    for (int it = 0; it < 32; ++it) {
      const int idx  = it * 32 + lane;   // 0..1023 = 16 nodes x 64 channels
      const int node = idx >> 6;
      const int ch   = idx & 63;
      const int gn   = row0 + node;
      const float cntv = C[gn];
      const float cc   = fmaxf(cntv, 1.0f);
      const float sv   = S[(size_t)gn * CF + ch];
      const float mean = sv / cc;
      const float var  = fmaxf(Q[(size_t)gn * CF + ch] / cc - mean * mean, 0.0f);
      const float stdv = sqrtf(var + EPSV);
      const float mxv  = cntv > 0.0f ? M[(size_t)gn * CF + ch] : 0.0f;
      _Float16* f = &sfeat[wid][node * 256];
      f[ch]        = (_Float16)sv;
      f[64 + ch]   = (_Float16)mean;
      f[128 + ch]  = (_Float16)mxv;
      f[192 + ch]  = (_Float16)stdv;
    }
    __syncthreads();

    // feats[16,256] @ W[256,64], accumulated across both aggregations.
    for (int kt = 0; kt < 8; ++kt) {
      v16h a;
      const _Float16* fb = &sfeat[wid][0];
#pragma unroll
      for (int v = 0; v < 8; ++v) {
        const int kl = (v < 4 ? 2 * v : 16 + 2 * (v - 4)) + akb;
        const int k  = kt * 32 + kl;
        a[2 * v]     = fb[am * 256 + k];
        a[2 * v + 1] = fb[am * 256 + k + 1];
      }
#pragma unroll
      for (int nt = 0; nt < 4; ++nt) {
        const v16h b =
            *(const v16h*)(Wh + ((size_t)(kt * 4 + nt) * 32 + lane) * 16);
        acc[nt] = wmma16(a, b, acc[nt]);
      }
    }
    __syncthreads();   // safe to overwrite sfeat for next aggregation
  }

  // bias + relu into LDS result tile
#pragma unroll
  for (int nt = 0; nt < 4; ++nt) {
    const int ncol = nt * 16 + bn;
    const float bias = b1[ncol] + b2[ncol];
#pragma unroll
    for (int v = 0; v < 8; ++v) {
      const int mm = v + (lane >> 4) * 8;
      sres[wid][mm * 64 + ncol] = fmaxf(acc[nt][v] + bias, 0.0f);
    }
  }
  __syncthreads();

  // LayerNorm over 64 channels; lanes 0..15 each own one node.
  if (lane < 16) {
    const float* r = &sres[wid][lane * 64];
    float mu = 0.f;
    for (int ch = 0; ch < 64; ++ch) mu += r[ch];
    mu *= (1.0f / 64.0f);
    float var = 0.f;
    for (int ch = 0; ch < 64; ++ch) { const float d = r[ch] - mu; var += d * d; }
    var *= (1.0f / 64.0f);
    const float rstd = rsqrtf(var + EPSV);
    float* o = out + (size_t)(row0 + lane) * CF;
    for (int ch = 0; ch < 64; ++ch)
      o[ch] = (r[ch] - mu) * rstd * gamma[ch] + beta[ch];
  }
}

// ---------------------------------------------------------------------------
extern "C" void kernel_launch(void* const* d_in, const int* in_sizes, int n_in,
                              void* d_out, int out_size, void* d_ws,
                              size_t ws_size, hipStream_t stream) {
  const float* x     = (const float*)d_in[0];
  const float* w1    = (const float*)d_in[1];
  const float* w2    = (const float*)d_in[2];
  const float* pw1   = (const float*)d_in[3];
  const float* pb1   = (const float*)d_in[4];
  const float* pw2   = (const float*)d_in[5];
  const float* pb2   = (const float*)d_in[6];
  const float* gamma = (const float*)d_in[7];
  const float* beta  = (const float*)d_in[8];
  const int* nbr = (const int*)d_in[9];
  const int* nbc = (const int*)d_in[10];
  const int* ccr = (const int*)d_in[11];
  const int* ccc = (const int*)d_in[12];
  float* out = (float*)d_out;

  // Workspace layout (floats): 2 msg + 2x(sum,sumsq,max) + 2 counts,
  // then f16 pre-packed weight fragments (32B-aligned).
  float* ws = (float*)d_ws;
  const size_t NC = (size_t)NODES * CF;
  float* msg1 = ws;
  float* msg2 = msg1 + NC;
  float* s1 = msg2 + NC;  float* q1 = s1 + NC;  float* m1 = q1 + NC;
  float* s2 = m1 + NC;    float* q2 = s2 + NC;  float* m2 = q2 + NC;
  float* c1 = m2 + NC;    float* c2 = c1 + NODES;
  _Float16* hbase = (_Float16*)(c2 + NODES + 14);  // keep 32B alignment slack
  hbase = (_Float16*)(((uintptr_t)hbase + 31) & ~(uintptr_t)31);
  _Float16* wh1  = hbase;                 // 64*64
  _Float16* wh2  = wh1 + 64 * 64;
  _Float16* pwh1 = wh2 + 64 * 64;         // 256*64
  _Float16* pwh2 = pwh1 + 256 * 64;

  const dim3 b256(256);
  const dim3 b128(128);

  // One-time per call: pack weights into WMMA B-fragment order (f16).
  pna_pack_b<<<2, b128, 0, stream>>>(w1, wh1, 64);     // 8 tiles * 32 lanes
  pna_pack_b<<<2, b128, 0, stream>>>(w2, wh2, 64);
  pna_pack_b<<<8, b128, 0, stream>>>(pw1, pwh1, 256);  // 32 tiles * 32 lanes
  pna_pack_b<<<8, b128, 0, stream>>>(pw2, pwh2, 256);

  const int initGrid = (NODES * CF + 255) / 256;
  pna_init_stats<<<initGrid, b256, 0, stream>>>(s1, q1, m1, c1);
  pna_init_stats<<<initGrid, b256, 0, stream>>>(s2, q2, m2, c2);

  pna_msg_gemm<<<(NTILES + 7) / 8, b256, 0, stream>>>(x, wh1, wh2, msg1, msg2);

  const int nthreads = 16 * ((NEDGE + EPT - 1) / EPT);
  const int edgeGrid = (nthreads + 255) / 256;
  pna_edge_stats<<<edgeGrid, b256, 0, stream>>>(nbr, nbc, msg1, s1, q1, m1, c1);
  pna_edge_stats<<<edgeGrid, b256, 0, stream>>>(ccr, ccc, msg2, s2, q2, m2, c2);

  pna_finalize<<<(NTILES + 3) / 4, b128, 0, stream>>>(
      s1, q1, m1, c1, s2, q2, m2, c2, pwh1, pb1, pwh2, pb2, gamma, beta, out);
}
